// SelfAttention_69595650065062
// MI455X (gfx1250) — matrix operand order repaired and checked by reading, not compiled, not run on hardware
//
#include <hip/hip_runtime.h>
#include <hip/hip_bf16.h>

typedef float    v2f  __attribute__((ext_vector_type(2)));
typedef float    v8f  __attribute__((ext_vector_type(8)));
typedef _Float16 v4h  __attribute__((ext_vector_type(4)));
typedef _Float16 v8h  __attribute__((ext_vector_type(8)));
typedef _Float16 v16h __attribute__((ext_vector_type(16)));
typedef uint32_t u32x4 __attribute__((ext_vector_type(4)));
typedef uint32_t u32x8 __attribute__((ext_vector_type(8)));

#define BDIM 4
#define LDIM 4096
#define DDIM 256
#define WAVES 2          // waves per block in attn kernel
#define KC 32            // keys per chunk
#define NCHUNK (LDIM / KC)
#define KB_STRIDE 260    // f32 key rows: 260 % 64 = 4 -> conflict-free column reads
#define XH_STRIDE 264    // f16 rows: 528B, 16B-aligned, 16 rows hit distinct bank groups
#define PS_STRIDE 36     // f32 P rows: 36*hl mod 64 distinct, 8B aligned

__device__ __forceinline__ v8f wmma_f32(v2f a, v2f b, v8f c) {
  return __builtin_amdgcn_wmma_f32_16x16x4_f32(false, a, false, b, (short)0, c, false, false);
}
__device__ __forceinline__ v8f wmma_f16(v16h a, v16h b, v8f c) {
  return __builtin_amdgcn_wmma_f32_16x16x32_f16(false, a, false, b, (short)0, c, false, false);
}
__device__ __forceinline__ v16h cat8(v8h a, v8h b) {
  return __builtin_shufflevector(a, b, 0,1,2,3,4,5,6,7,8,9,10,11,12,13,14,15);
}
__device__ __forceinline__ void wait_tensorcnt0() {
#if __has_builtin(__builtin_amdgcn_s_wait_tensorcnt)
  __builtin_amdgcn_s_wait_tensorcnt(0);
#else
  asm volatile("s_wait_tensorcnt 0x0" ::: "memory");
#endif
}

// TDM: async DMA one 32x256 f32 tile (row stride 256 f32) global -> LDS,
// LDS rows padded by TDM to 260 dwords (pad_interval=256dw, pad_amount=4dw).
__device__ __forceinline__ void tdm_load_tile(const float* gsrc, void* lds_dst) {
  uint64_t ga  = (uint64_t)(uintptr_t)gsrc;
  uint32_t lds = (uint32_t)(uintptr_t)lds_dst;   // low 32 bits of generic LDS ptr = LDS offset
  u32x4 g0;
  g0[0] = 1u;                                            // count=1, no gather
  g0[1] = lds;                                           // lds_addr
  g0[2] = (uint32_t)ga;                                  // global_addr[31:0]
  g0[3] = ((uint32_t)(ga >> 32) & 0x01FFFFFFu) | 0x80000000u; // addr[56:32] | type=2
  u32x8 g1;
  g1[0] = (2u << 16) | (1u << 20) | (7u << 22) | (3u << 25);  // data_size=4B, pad: +4dw/256dw
  g1[1] = (256u << 16);            // tensor_dim0 = 256 elements (bits 79:48, low part)
  g1[2] = (KC << 16);              // tensor_dim0 hi=0 | tensor_dim1 = 32 (low 16)
  g1[3] = (256u << 16);            // tensor_dim1 hi=0 | tile_dim0 = 256
  g1[4] = KC;                      // tile_dim1 = 32, tile_dim2 = 0
  g1[5] = 256u;                    // tensor_dim0_stride = 256 elements
  g1[6] = 0u;
  g1[7] = 0u;                      // tensor_dim1_stride unused (2D)
  asm volatile("tensor_load_to_lds %0, %1" :: "s"(g0), "s"(g1) : "memory");
}

// ---------------- Kernel 1: Q = x @ W (f32 WMMA, 16x16 tile per wave) ------------
__global__ __launch_bounds__(256) void qproj_kernel(const float* __restrict__ x,
                                                    const float* __restrict__ W,
                                                    float* __restrict__ Q) {
  const int lane = threadIdx.x & 31;
  const int wv   = threadIdx.x >> 5;
  const int half = lane >> 4;
  const int hl   = lane & 15;
  const int gw   = blockIdx.x * 8 + wv;
  const int mtile = gw >> 4;
  const int ntile = gw & 15;
  const int row   = mtile * 16 + hl;
  const int ncol  = ntile * 16 + hl;

  v8f acc = {0.f,0.f,0.f,0.f,0.f,0.f,0.f,0.f};
  const float* xr = x + (size_t)row * DDIM;
  #pragma unroll 8
  for (int kb = 0; kb < DDIM; kb += 4) {
    v2f a = *(const v2f*)(xr + kb + 2 * half);
    v2f b;
    b.x = W[(size_t)(kb + 2 * half + 0) * DDIM + ncol];
    b.y = W[(size_t)(kb + 2 * half + 1) * DDIM + ncol];
    acc = wmma_f32(a, b, acc);
  }
  #pragma unroll
  for (int v = 0; v < 8; ++v)
    Q[(size_t)(mtile * 16 + v + 8 * half) * DDIM + ntile * 16 + hl] = acc[v];
}

// ---------------- Kernel 2: flash attention -----------------------------------
// Scores: split-f16 (hi/lo) 3-term WMMA f32_16x16x32_f16; P@X: f32 WMMA.
// Key tiles staged by TDM (tensor_load_to_lds), double-buffered.
__global__ __launch_bounds__(64) void attn_kernel(const float* __restrict__ x,
                                                  const float* __restrict__ Q,
                                                  float* __restrict__ out) {
  __shared__ float    kbuf[2][KC][KB_STRIDE];     // raw f32 key tiles from TDM
  __shared__ _Float16 xkhi[KC][XH_STRIDE];        // f16 split of current key tile
  __shared__ _Float16 xklo[KC][XH_STRIDE];
  __shared__ _Float16 qhi[WAVES][16][XH_STRIDE];  // f16 split of Q tiles
  __shared__ _Float16 qlo[WAVES][16][XH_STRIDE];
  __shared__ float    ps[WAVES][16][PS_STRIDE];   // P tile (C -> A relayout)

  const int tid  = threadIdx.x;
  const int lane = tid & 31;
  const int wv   = tid >> 5;
  const int half = lane >> 4;
  const int hl   = lane & 15;

  const int qblocks_per_batch = LDIM / (16 * WAVES);
  const int batch = blockIdx.x / qblocks_per_batch;
  const int qblk  = blockIdx.x % qblocks_per_batch;
  const int qrow0 = batch * LDIM + qblk * (16 * WAVES) + wv * 16;
  const float* xb = x + (size_t)batch * LDIM * DDIM;

  // Kick off TDM for chunk 0, overlapping with Q staging below.
  if (wv == 0) tdm_load_tile(xb, &kbuf[0][0][0]);

  // Stage this wave's Q tile: global f32 -> LDS f16 hi/lo split.
  {
    const float* qg = Q + (size_t)qrow0 * DDIM;
    #pragma unroll
    for (int r = 0; r < 16; ++r) {
      #pragma unroll
      for (int c4 = 0; c4 < 8; c4 += 4) {
        float4 v = *(const float4*)(qg + r * DDIM + lane * 8 + c4);
        v4h hi, lo;
        hi[0] = (_Float16)v.x; lo[0] = (_Float16)(v.x - (float)hi[0]);
        hi[1] = (_Float16)v.y; lo[1] = (_Float16)(v.y - (float)hi[1]);
        hi[2] = (_Float16)v.z; lo[2] = (_Float16)(v.z - (float)hi[2]);
        hi[3] = (_Float16)v.w; lo[3] = (_Float16)(v.w - (float)hi[3]);
        *(v4h*)&qhi[wv][r][lane * 8 + c4] = hi;
        *(v4h*)&qlo[wv][r][lane * 8 + c4] = lo;
      }
    }
  }

  float m_i[8], l_i[8];
  #pragma unroll
  for (int v = 0; v < 8; ++v) { m_i[v] = -INFINITY; l_i[v] = 0.f; }
  v8f o[16];
  #pragma unroll
  for (int t = 0; t < 16; ++t) o[t] = (v8f){0.f,0.f,0.f,0.f,0.f,0.f,0.f,0.f};

  const int cr = tid >> 1;          // conversion: row 0..31
  const int ch = (tid & 1) * 128;   // conversion: half-row offset

  for (int kt = 0; kt < NCHUNK; ++kt) {
    const int cur = kt & 1;
    if (wv == 0) wait_tensorcnt0();      // TDM(kt) landed in kbuf[cur]
    __syncthreads();                     // kbuf[cur] visible; prev chunk fully consumed
    if (wv == 0 && kt + 1 < NCHUNK)      // prefetch next chunk into the free buffer
      tdm_load_tile(xb + (size_t)(kt + 1) * KC * DDIM, &kbuf[1 - cur][0][0]);

    // Convert current key tile f32 -> f16 hi/lo (64 threads, 128 elems each)
    #pragma unroll 8
    for (int i = 0; i < 128; i += 4) {
      float4 v = *(const float4*)&kbuf[cur][cr][ch + i];
      v4h hi, lo;
      hi[0] = (_Float16)v.x; lo[0] = (_Float16)(v.x - (float)hi[0]);
      hi[1] = (_Float16)v.y; lo[1] = (_Float16)(v.y - (float)hi[1]);
      hi[2] = (_Float16)v.z; lo[2] = (_Float16)(v.z - (float)hi[2]);
      hi[3] = (_Float16)v.w; lo[3] = (_Float16)(v.w - (float)hi[3]);
      *(v4h*)&xkhi[cr][ch + i] = hi;
      *(v4h*)&xklo[cr][ch + i] = lo;
    }
    __syncthreads();                     // f16 key tiles ready

    // ---- S(16x32) = Qw @ Xk^T via split-f16: per K=32 block, 3 WMMAs/subtile
    v8f s0 = (v8f){0.f,0.f,0.f,0.f,0.f,0.f,0.f,0.f};
    v8f s1 = (v8f){0.f,0.f,0.f,0.f,0.f,0.f,0.f,0.f};
    #pragma unroll 2
    for (int kb = 0; kb < DDIM; kb += 32) {
      // A lane(m=hl,half): K in {kb+16h..+15} -> two 16B loads
      v16h ahi = cat8(*(const v8h*)&qhi[wv][hl][kb + 16 * half],
                      *(const v8h*)&qhi[wv][hl][kb + 16 * half + 8]);
      v16h alo = cat8(*(const v8h*)&qlo[wv][hl][kb + 16 * half],
                      *(const v8h*)&qlo[wv][hl][kb + 16 * half + 8]);
      // B lane(n=hl,half): Xk[key][kb+16*half .. +15] -> two 16B loads
      v16h b0h = cat8(*(const v8h*)&xkhi[hl][kb + 16 * half],
                      *(const v8h*)&xkhi[hl][kb + 16 * half + 8]);
      v16h b0l = cat8(*(const v8h*)&xklo[hl][kb + 16 * half],
                      *(const v8h*)&xklo[hl][kb + 16 * half + 8]);
      v16h b1h = cat8(*(const v8h*)&xkhi[16 + hl][kb + 16 * half],
                      *(const v8h*)&xkhi[16 + hl][kb + 16 * half + 8]);
      v16h b1l = cat8(*(const v8h*)&xklo[16 + hl][kb + 16 * half],
                      *(const v8h*)&xklo[16 + hl][kb + 16 * half + 8]);
      s0 = wmma_f16(ahi, b0h, s0);
      s0 = wmma_f16(ahi, b0l, s0);
      s0 = wmma_f16(alo, b0h, s0);
      s1 = wmma_f16(ahi, b1h, s1);
      s1 = wmma_f16(ahi, b1l, s1);
      s1 = wmma_f16(alo, b1h, s1);
    }

    // ---- online softmax over 32 keys (rows live in 16-lane halves)
    float sc[8];
    #pragma unroll
    for (int v = 0; v < 8; ++v) {
      float rm = fmaxf(s0[v], s1[v]);
      rm = fmaxf(rm, __shfl_xor(rm, 1, 32));
      rm = fmaxf(rm, __shfl_xor(rm, 2, 32));
      rm = fmaxf(rm, __shfl_xor(rm, 4, 32));
      rm = fmaxf(rm, __shfl_xor(rm, 8, 32));
      float mnew = fmaxf(m_i[v], rm);
      sc[v] = __expf(m_i[v] - mnew);
      float p0 = __expf(s0[v] - mnew);
      float p1 = __expf(s1[v] - mnew);
      float rs = p0 + p1;
      rs += __shfl_xor(rs, 1, 32);
      rs += __shfl_xor(rs, 2, 32);
      rs += __shfl_xor(rs, 4, 32);
      rs += __shfl_xor(rs, 8, 32);
      l_i[v] = l_i[v] * sc[v] + rs;
      m_i[v] = mnew;
      ps[wv][v + 8 * half][hl]      = p0;
      ps[wv][v + 8 * half][16 + hl] = p1;
    }
    #pragma unroll
    for (int t = 0; t < 16; ++t)
      #pragma unroll
      for (int v = 0; v < 8; ++v) o[t][v] *= sc[v];

    __syncthreads();                     // ps visible (uniform barrier)

    // ---- O(16x256) += P(16x32) @ Xk(32x256) in f32 WMMA (exact, key-major B)
    v2f pa[8];
    #pragma unroll
    for (int kk = 0; kk < 8; ++kk)
      pa[kk] = *(const v2f*)&ps[wv][hl][kk * 4 + 2 * half];
    #pragma unroll
    for (int t = 0; t < 16; ++t) {
      const int cn = t * 16 + hl;
      #pragma unroll
      for (int kk = 0; kk < 8; ++kk) {
        v2f b;
        b.x = kbuf[cur][kk * 4 + 2 * half + 0][cn];
        b.y = kbuf[cur][kk * 4 + 2 * half + 1][cn];
        o[t] = wmma_f32(pa[kk], b, o[t]);
      }
    }
  }

  // Epilogue: O /= l
  float invl[8];
  #pragma unroll
  for (int v = 0; v < 8; ++v) invl[v] = 1.0f / l_i[v];
  #pragma unroll
  for (int t = 0; t < 16; ++t)
    #pragma unroll
    for (int v = 0; v < 8; ++v)
      out[(size_t)(qrow0 + v + 8 * half) * DDIM + t * 16 + hl] = o[t][v] * invl[v];
}

extern "C" void kernel_launch(void* const* d_in, const int* in_sizes, int n_in,
                              void* d_out, int out_size, void* d_ws, size_t ws_size,
                              hipStream_t stream) {
  const float* x = (const float*)d_in[0];
  const float* W = (const float*)d_in[1];
  float* out = (float*)d_out;
  float* Qw  = (float*)d_ws;

  qproj_kernel<<<dim3((BDIM * LDIM / 16) * (DDIM / 16) / 8), dim3(256), 0, stream>>>(x, W, Qw);
  attn_kernel<<<dim3(BDIM * (LDIM / (16 * WAVES))), dim3(64), 0, stream>>>(x, Qw, out);
}
